// SelfAttentionMamba_39324720562851
// MI455X (gfx1250) — matrix operand order repaired and checked by reading, not compiled, not run on hardware
//
#include <hip/hip_runtime.h>
#include <hip/hip_bf16.h>
#include <math.h>

typedef _Float16 v16h __attribute__((ext_vector_type(16)));
typedef _Float16 v8h  __attribute__((ext_vector_type(8)));
typedef float    v8f  __attribute__((ext_vector_type(8)));

#define BQ   64
#define SEQ  512
#define DIM  512
#define HID  256
#define SCALE_QK 0.044194173824159216f   // 1/sqrt(512)

// epilogue modes
#define EPI_F16    0   // store f16 row-major
#define EPI_VBOTH  1   // store f16 row-major (Ch) + transposed per batch (Ch2)
#define EPI_F32    2   // store f32 (optional bias)
#define EPI_GELU16 3   // bias + exact GELU, store f16
#define EPI_SCORES 4   // *scale + rel_bias + causal mask, store f32

// ---------- CDNA5 async global->LDS copy (ASYNCcnt-tracked) ----------
__device__ __forceinline__ void async_ld_b128(unsigned lds_byte_off, const void* gaddr) {
  asm volatile("global_load_async_to_lds_b128 %0, %1, off"
               :: "v"(lds_byte_off), "v"(gaddr) : "memory");
}
__device__ __forceinline__ void wait_async_le4() {
  asm volatile("s_wait_asynccnt 0x4" ::: "memory");
}
__device__ __forceinline__ void wait_async_le0() {
  asm volatile("s_wait_asynccnt 0x0" ::: "memory");
}

__device__ __forceinline__ v16h ld_frag(const _Float16* p0, const _Float16* p1) {
  v8h lo = *(const v8h*)p0;
  v8h hi = *(const v8h*)p1;
  v16h r;
#pragma unroll
  for (int j = 0; j < 8; ++j) { r[j] = lo[j]; r[j + 8] = hi[j]; }
  return r;
}

// C[M,N] = A[M,K] * B^T where B is stored N x K row-major ("NT" form).
// Block: 256 threads = 8 waves; block tile 128x128; wave tile 64x32; K-step 32.
// Double-buffered LDS filled with global_load_async_to_lds_b128; last stage peeled.
// kmode==1: bound K at tileM+128 (P@V: P[q,k]==0 for k>q).
template<int EPI>
__global__ __launch_bounds__(256)
void gemm_wmma_nt(const _Float16* __restrict__ Ag, const _Float16* __restrict__ Bg,
                  const float* __restrict__ bias, const float* __restrict__ rel,
                  float* Cf, _Float16* Ch, _Float16* Ch2,
                  int M, int N, int K, long sA, long sB, long sC, int kmode)
{
  constexpr int LD   = 40;                 // padded halves: 80B rows, keeps 16B align
  constexpr int AHLF = 128 * LD;           // halves per A tile
  constexpr int BUFH = 2 * AHLF;           // halves per (A+B) buffer
  __shared__ __align__(16) _Float16 smem[2 * BUFH];  // 40960 B, double buffered

  const int tid   = threadIdx.x;
  const int wid   = tid >> 5;
  const int lane  = tid & 31;
  const int hgrp  = lane >> 4;             // 0: lanes 0-15, 1: lanes 16-31
  const int lr    = lane & 15;
  const int waveM = wid & 1;               // 2 waves along M (64 rows each)
  const int waveN = wid >> 1;              // 4 waves along N (32 cols each)
  const int tileM = blockIdx.y * 128;
  const int tileN = blockIdx.x * 128;
  const long bz   = blockIdx.z;

  // fully causal-masked scores tile: fill -inf and leave
  if constexpr (EPI == EPI_SCORES) {
    if (tileN > tileM + 127) {
      const float ninf = -__builtin_inff();
      float4 f4 = make_float4(ninf, ninf, ninf, ninf);
      for (int i = tid; i < (128 * 128) / 4; i += 256) {
        int idx = i * 4;
        int row = idx >> 7;
        int col = idx & 127;
        *(float4*)&Cf[bz * sC + (long)(tileM + row) * N + tileN + col] = f4;
      }
      return;
    }
  }

  const _Float16* A = Ag + bz * sA;
  const _Float16* B = Bg + bz * sB;

  const int Kend = (kmode == 1) ? min(K, tileM + 128) : K;

  // staging coordinates: 128 rows x 32 halves per tile, 16B per op, 2 ops per matrix
  const int srow = tid >> 2;               // 0..63  (+64 on second op)
  const int sc8  = (tid & 3) * 8;          // 0,8,16,24

  // per-thread source pointers, advanced 32 halves per stage
  const _Float16* pa0 = A + (long)(tileM + srow     ) * K + sc8;
  const _Float16* pa1 = A + (long)(tileM + srow + 64) * K + sc8;
  const _Float16* pb0 = B + (long)(tileN + srow     ) * K + sc8;
  const _Float16* pb1 = B + (long)(tileN + srow + 64) * K + sc8;
  const unsigned la0 = (unsigned)((srow      ) * LD + sc8) * 2u;
  const unsigned la1 = (unsigned)((srow + 64 ) * LD + sc8) * 2u;
  const unsigned lb0 = la0 + AHLF * 2u;
  const unsigned lb1 = la1 + AHLF * 2u;

  auto issue = [&](int buf) {
    unsigned base = (unsigned)buf * (BUFH * 2u);
    async_ld_b128(base + la0, pa0);
    async_ld_b128(base + la1, pa1);
    async_ld_b128(base + lb0, pb0);
    async_ld_b128(base + lb1, pb1);
    pa0 += 32; pa1 += 32; pb0 += 32; pb1 += 32;
  };

  v8f vz = {0.f,0.f,0.f,0.f,0.f,0.f,0.f,0.f};
  v8f acc[4][2] = {{vz,vz},{vz,vz},{vz,vz},{vz,vz}};

  auto compute = [&](int buf) {
    const _Float16* As = smem + buf * BUFH;
    const _Float16* Bs = As + AHLF;
    v16h af[4], bf[2];
#pragma unroll
    for (int mi = 0; mi < 4; ++mi) {
      int row = waveM * 64 + mi * 16 + lr;
      int k0  = hgrp * 8;                  // lanes 0-15: K 0..7/16..23; 16-31: 8..15/24..31
      af[mi] = ld_frag(&As[row * LD + k0], &As[row * LD + k0 + 16]);
    }
#pragma unroll
    for (int ni = 0; ni < 2; ++ni) {
      int n  = waveN * 32 + ni * 16 + lr;
      int kb = hgrp * 8;                   // lanes 0-15: K 0..7/16..23; 16-31: 8..15/24..31
      bf[ni] = ld_frag(&Bs[n * LD + kb], &Bs[n * LD + kb + 16]);
    }
#pragma unroll
    for (int mi = 0; mi < 4; ++mi)
#pragma unroll
      for (int ni = 0; ni < 2; ++ni)
        acc[mi][ni] = __builtin_amdgcn_wmma_f32_16x16x32_f16(
            false, af[mi], false, bf[ni], (short)0, acc[mi][ni], false, false);
  };

  // software pipeline: steady state always has the next stage in flight
  issue(0);
  int cur = 0;
  for (int kt = 0; kt < Kend - 32; kt += 32) {
    issue(cur ^ 1);
    wait_async_le4();                      // stage for buf[cur] has landed
    __syncthreads();
    compute(cur);
    __syncthreads();                       // everyone done reading buf[cur]
    cur ^= 1;
  }
  wait_async_le0();
  __syncthreads();
  compute(cur);

  // ---- epilogue (C layout: VGPR r -> row mi*16 + hgrp*8 + r, col = lr)
#pragma unroll
  for (int mi = 0; mi < 4; ++mi) {
#pragma unroll
    for (int ni = 0; ni < 2; ++ni) {
#pragma unroll
      for (int r = 0; r < 8; ++r) {
        int row = tileM + waveM * 64 + mi * 16 + hgrp * 8 + r;
        int col = tileN + waveN * 32 + ni * 16 + lr;
        float val = acc[mi][ni][r];
        if constexpr (EPI == EPI_SCORES) {
          float s = val * SCALE_QK + rel[(long)row * SEQ + col];
          if (col > row) s = -__builtin_inff();
          Cf[bz * sC + (long)row * N + col] = s;
        } else if constexpr (EPI == EPI_VBOTH) {
          _Float16 h = (_Float16)val;
          Ch[(long)row * N + col] = h;                    // normal layout (conv)
          long b = row >> 9, n = row & 511;
          Ch2[(b * DIM + col) * SEQ + n] = h;             // transposed (P@V B operand)
        } else {
          if (bias) val += bias[col];
          if constexpr (EPI == EPI_GELU16)
            val = 0.5f * val * (1.0f + erff(val * 0.70710678118654752f));
          long cidx = bz * sC + (long)row * N + col;
          if constexpr (EPI == EPI_F32) Cf[cidx] = val;
          else                          Ch[cidx] = (_Float16)val;
        }
      }
    }
  }
}

__global__ void cvt_f32_to_f16(const float* __restrict__ s, _Float16* __restrict__ d, long n) {
  long i  = (long)blockIdx.x * blockDim.x + threadIdx.x;
  long st = (long)gridDim.x * blockDim.x;
  for (; i < n; i += st) d[i] = (_Float16)s[i];
}

// out[c*R + r] = (f16)in[r*C + c]   (weight transpose to N x K form)
__global__ void cvt_transpose_f16(const float* __restrict__ in, _Float16* __restrict__ out,
                                  int R, int C) {
  long i  = (long)blockIdx.x * blockDim.x + threadIdx.x;
  long n  = (long)R * C;
  long st = (long)gridDim.x * blockDim.x;
  for (; i < n; i += st) {
    long r = i / C, c = i % C;
    out[c * (long)R + r] = (_Float16)in[i];
  }
}

// one wave per 512-entry row; 4 rows per 128-thread block
__global__ __launch_bounds__(128)
void softmax_rows(const float* __restrict__ S, _Float16* __restrict__ P) {
  long row = (long)blockIdx.x * 4 + (threadIdx.x >> 5);
  int lane = threadIdx.x & 31;
  const float* r = S + row * SEQ;
  float v[16];
  float m = -__builtin_inff();
#pragma unroll
  for (int j = 0; j < 16; ++j) { v[j] = r[lane + 32 * j]; m = fmaxf(m, v[j]); }
#pragma unroll
  for (int off = 16; off >= 1; off >>= 1) m = fmaxf(m, __shfl_xor(m, off, 32));
  float s = 0.f;
#pragma unroll
  for (int j = 0; j < 16; ++j) { v[j] = expf(v[j] - m); s += v[j]; }
#pragma unroll
  for (int off = 16; off >= 1; off >>= 1) s += __shfl_xor(s, off, 32);
  float inv = 1.f / s;
  _Float16* p = P + row * SEQ;
#pragma unroll
  for (int j = 0; j < 16; ++j) p[lane + 32 * j] = (_Float16)(v[j] * inv);
}

// out = att + depthwise_conv3(v) + conv_b, in place on att; also f16 copy.
__global__ __launch_bounds__(256)
void conv_add_rows(float* att, const _Float16* __restrict__ v,
                   const float* __restrict__ cw, const float* __restrict__ cb,
                   _Float16* __restrict__ outh) {
  long rowid = blockIdx.x;                // 0 .. B*SEQ-1
  int  b = (int)(rowid >> 9), n = (int)(rowid & 511);
  const _Float16* vb = v + ((long)b * SEQ) * DIM;
  for (int d = threadIdx.x; d < DIM; d += 256) {
    float w0 = cw[d * 3], w1 = cw[d * 3 + 1], w2 = cw[d * 3 + 2];
    float vm = (n > 0)   ? (float)vb[(long)(n - 1) * DIM + d] : 0.f;
    float v0 =             (float)vb[(long)n       * DIM + d];
    float vp = (n < 511) ? (float)vb[(long)(n + 1) * DIM + d] : 0.f;
    long idx = rowid * DIM + d;
    float o = att[idx] + (w0 * vm + w1 * v0 + w2 * vp + cb[d]);
    att[idx]  = o;
    outh[idx] = (_Float16)o;
  }
}

// t = x + sigmoid(gate_lin) * out ; LayerNorm over 512 feats -> f16
__global__ __launch_bounds__(256)
void gate_ln_rows(const float* __restrict__ x, const float* __restrict__ outf,
                  const float* __restrict__ gl, const float* __restrict__ lng,
                  const float* __restrict__ lnb, _Float16* __restrict__ nh) {
  __shared__ float s1[8], s2[8];
  long row = blockIdx.x;
  int tid = threadIdx.x, wid = tid >> 5, lane = tid & 31;
  int dd[2] = {tid, tid + 256};
  float t[2]; float lsum = 0.f, lsq = 0.f;
#pragma unroll
  for (int i = 0; i < 2; ++i) {
    long idx = row * DIM + dd[i];
    float o = outf[idx];
    float g = gl[idx];
    float sg = 1.f / (1.f + expf(-g));
    float tt = x[idx] + sg * o;
    t[i] = tt; lsum += tt; lsq += tt * tt;
  }
#pragma unroll
  for (int off = 16; off >= 1; off >>= 1) {
    lsum += __shfl_xor(lsum, off, 32);
    lsq  += __shfl_xor(lsq,  off, 32);
  }
  if (lane == 0) { s1[wid] = lsum; s2[wid] = lsq; }
  __syncthreads();
  float tot = 0.f, tot2 = 0.f;
#pragma unroll
  for (int w = 0; w < 8; ++w) { tot += s1[w]; tot2 += s2[w]; }
  float mu  = tot * (1.f / DIM);
  float var = tot2 * (1.f / DIM) - mu * mu;
  float rstd = rsqrtf(var + 1e-5f);
#pragma unroll
  for (int i = 0; i < 2; ++i) {
    float nn = (t[i] - mu) * rstd * lng[dd[i]] + lnb[dd[i]];
    nh[row * DIM + dd[i]] = (_Float16)nn;
  }
}

extern "C" void kernel_launch(void* const* d_in, const int* in_sizes, int n_in,
                              void* d_out, int out_size, void* d_ws, size_t ws_size,
                              hipStream_t stream) {
  const float* x      = (const float*)d_in[0];
  const float* Wq     = (const float*)d_in[1];
  const float* Wk     = (const float*)d_in[2];
  const float* Wv     = (const float*)d_in[3];
  const float* rel    = (const float*)d_in[4];
  const float* conv_w = (const float*)d_in[5];
  const float* conv_b = (const float*)d_in[6];
  const float* gate_w = (const float*)d_in[7];
  const float* gate_b = (const float*)d_in[8];
  const float* ln_g   = (const float*)d_in[9];
  const float* ln_b   = (const float*)d_in[10];
  const float* fc1_w  = (const float*)d_in[11];
  const float* fc1_b  = (const float*)d_in[12];
  const float* fc2_w  = (const float*)d_in[13];
  const float* fc2_b  = (const float*)d_in[14];

  const long NT = (long)BQ * SEQ;        // 32768 rows
  const long XE = NT * DIM;              // 16,777,216 elems
  const long SE = (long)BQ * SEQ * SEQ;  // scores elems
  const long BS = (long)SEQ * SEQ;       // per-batch stride

  char* ws = (char*)d_ws;
  size_t off = 0;
  auto take = [&](size_t bytes) -> char* {
    char* p = ws + off;
    off += (bytes + 255) & ~(size_t)255;
    return p;
  };
  _Float16* xh  = (_Float16*)take(XE * 2);          // later reused as out_h
  _Float16* wqT = (_Float16*)take((size_t)DIM * DIM * 2);
  _Float16* wkT = (_Float16*)take((size_t)DIM * DIM * 2);
  _Float16* wvT = (_Float16*)take((size_t)DIM * DIM * 2);
  _Float16* gwT = (_Float16*)take((size_t)DIM * DIM * 2);
  _Float16* f1T = (_Float16*)take((size_t)DIM * HID * 2);
  _Float16* f2T = (_Float16*)take((size_t)HID * DIM * 2);
  _Float16* qh  = (_Float16*)take(XE * 2);          // later reused as MLP hidden
  _Float16* kh  = (_Float16*)take(XE * 2);
  _Float16* vh  = (_Float16*)take(XE * 2);          // V normal layout (conv)
  _Float16* vT  = (_Float16*)take(XE * 2);          // V transposed per batch (P@V)
  float*    S   = (float*)   take(SE * 4);          // later reused as gate_lin
  _Float16* Ph  = (_Float16*)take(SE * 2);          // later reused as normed f16
  float*    att = (float*)   take(SE * 4);          // becomes `out` in place

  _Float16* outh = xh;
  float*    gl   = S;
  _Float16* nh   = Ph;
  _Float16* hh   = qh;

  // conversions (weights transposed to N x K form; A operands row-major)
  cvt_f32_to_f16<<<4096, 256, 0, stream>>>(x, xh, XE);
  cvt_transpose_f16<<<512, 256, 0, stream>>>(Wq, wqT, DIM, DIM);
  cvt_transpose_f16<<<512, 256, 0, stream>>>(Wk, wkT, DIM, DIM);
  cvt_transpose_f16<<<512, 256, 0, stream>>>(Wv, wvT, DIM, DIM);
  cvt_transpose_f16<<<512, 256, 0, stream>>>(gate_w, gwT, DIM, DIM);
  cvt_transpose_f16<<<256, 256, 0, stream>>>(fc1_w, f1T, DIM, HID);
  cvt_transpose_f16<<<256, 256, 0, stream>>>(fc2_w, f2T, HID, DIM);

  // QKV projections: [32768x512] x [512x512]^T -> f16 (V also stored transposed)
  dim3 gQKV(DIM / 128, NT / 128, 1);
  gemm_wmma_nt<EPI_F16><<<gQKV, 256, 0, stream>>>(
      xh, wqT, nullptr, nullptr, nullptr, qh, nullptr, (int)NT, DIM, DIM, 0, 0, 0, 0);
  gemm_wmma_nt<EPI_F16><<<gQKV, 256, 0, stream>>>(
      xh, wkT, nullptr, nullptr, nullptr, kh, nullptr, (int)NT, DIM, DIM, 0, 0, 0, 0);
  gemm_wmma_nt<EPI_VBOTH><<<gQKV, 256, 0, stream>>>(
      xh, wvT, nullptr, nullptr, nullptr, vh, vT, (int)NT, DIM, DIM, 0, 0, 0, 0);

  // scores: S = q k^T * scale + rel_bias, causal (masked tiles skipped entirely)
  dim3 gATT(SEQ / 128, SEQ / 128, BQ);
  gemm_wmma_nt<EPI_SCORES><<<gATT, 256, 0, stream>>>(
      qh, kh, nullptr, rel, S, nullptr, nullptr, SEQ, SEQ, DIM, BS, BS, BS, 0);

  // softmax rows -> P (f16)
  softmax_rows<<<(BQ * SEQ) / 4, 128, 0, stream>>>(S, Ph);

  // att_out = P @ V : B operand is vT; K bounded by causal structure (kmode=1)
  gemm_wmma_nt<EPI_F32><<<gATT, 256, 0, stream>>>(
      Ph, vT, nullptr, nullptr, att, nullptr, nullptr, SEQ, DIM, SEQ, BS, BS, BS, 1);

  // out = att_out + depthwise conv3(v) + conv_b (in place), plus f16 copy
  conv_add_rows<<<(int)NT, 256, 0, stream>>>(att, vh, conv_w, conv_b, outh);

  // gate_lin = out @ gate_w + gate_b -> f32 (reuses S buffer)
  gemm_wmma_nt<EPI_F32><<<gQKV, 256, 0, stream>>>(
      outh, gwT, gate_b, nullptr, gl, nullptr, nullptr, (int)NT, DIM, DIM, 0, 0, 0, 0);

  // gated residual + LayerNorm -> normed f16
  gate_ln_rows<<<(int)NT, 256, 0, stream>>>(x, att, gl, ln_g, ln_b, nh);

  // fc1 + exact GELU -> f16 hidden [32768 x 256]
  dim3 gFC1(HID / 128, NT / 128, 1);
  gemm_wmma_nt<EPI_GELU16><<<gFC1, 256, 0, stream>>>(
      nh, f1T, fc1_b, nullptr, nullptr, hh, nullptr, (int)NT, HID, DIM, 0, 0, 0, 0);

  // fc2 -> f32 output
  dim3 gFC2(DIM / 128, NT / 128, 1);
  gemm_wmma_nt<EPI_F32><<<gFC2, 256, 0, stream>>>(
      hh, f2T, fc2_b, nullptr, (float*)d_out, nullptr, nullptr, (int)NT, DIM, HID, 0, 0, 0, 0);

  (void)in_sizes; (void)n_in; (void)out_size; (void)ws_size;
}
